// GPTNeoModel_38062000177425
// MI455X (gfx1250) — matrix operand order repaired and checked by reading, not compiled, not run on hardware
//
#include <hip/hip_runtime.h>
#include <hip/hip_bf16.h>
#include <math.h>

// Model dims
#define S_LEN 1024
#define D_DIM 768
#define H_NUM 12
#define HD    64
#define F_DIM 3072
#define V_DIM 50257
#define L_NUM 12

typedef __attribute__((ext_vector_type(16))) _Float16 v16h;
typedef __attribute__((ext_vector_type(8)))  _Float16 v8h;
typedef __attribute__((ext_vector_type(8)))  float    v8f;

// ---- gfx1250 async global->LDS copy (ASYNCcnt path), with safe fallback ----
#if defined(__gfx1250__) && __has_builtin(__builtin_amdgcn_global_load_async_to_lds_b128)
#define HAVE_ASYNC_LDS 1
typedef int v4i_vs __attribute__((vector_size(16)));
typedef __attribute__((address_space(1))) v4i_vs* GlobalV4iPtr;
typedef __attribute__((address_space(3))) v4i_vs* LdsV4iPtr;
__device__ __forceinline__ void async_copy16(const void* gsrc, void* ldst) {
    __builtin_amdgcn_global_load_async_to_lds_b128((GlobalV4iPtr)gsrc, (LdsV4iPtr)ldst, 0, 0);
}
__device__ __forceinline__ void async_wait0() {
#if __has_builtin(__builtin_amdgcn_s_wait_asynccnt)
    __builtin_amdgcn_s_wait_asynccnt(0);
#else
    asm volatile("s_wait_asynccnt 0x0" ::: "memory");
#endif
}
#endif

__device__ __forceinline__ v16h cat8(v8h lo, v8h hi) {
    v16h r;
#pragma unroll
    for (int j = 0; j < 8; ++j) { r[j] = lo[j]; r[j + 8] = hi[j]; }
    return r;
}
__device__ __forceinline__ v16h ld16h(const _Float16* p) {
    return cat8(*(const v8h*)p, *(const v8h*)(p + 8));
}
__device__ __forceinline__ float gelu_exact(float x) {
    return 0.5f * x * (1.0f + erff(x * 0.70710678118654752f));
}

// ---------------------------------------------------------------------------
// Embedding + sinusoidal positional encoding: x[s,d] = wte[ids[s],d] + pe(s,d)
// ---------------------------------------------------------------------------
__global__ __launch_bounds__(256) void embed_kernel(const int* __restrict__ ids,
                                                    const float* __restrict__ wte,
                                                    float* __restrict__ X) {
    const int s  = blockIdx.x;
    const int id = ids[s];
    for (int d = threadIdx.x; d < D_DIM; d += 256) {
        const int   i   = d >> 1;
        const float ang = (float)s * __expf((float)(2 * i) * (-9.210340371976184f / (float)D_DIM));
        const float pe  = (d & 1) ? __cosf(ang) : __sinf(ang);
        X[(size_t)s * D_DIM + d] = wte[(size_t)id * D_DIM + d] + pe;
    }
}

// ---------------------------------------------------------------------------
// LayerNorm over D=768, one row per block, optional fused residual add.
// ---------------------------------------------------------------------------
__global__ __launch_bounds__(256) void ln_kernel(const float* __restrict__ X,
                                                 const float* __restrict__ Res,
                                                 const float* __restrict__ scale,
                                                 const float* __restrict__ bias,
                                                 float* __restrict__ Y) {
    __shared__ float sdata[256];
    const int    row = blockIdx.x;
    const float* x   = X + (size_t)row * D_DIM;
    float v[3];
    float s = 0.f;
#pragma unroll
    for (int j = 0; j < 3; ++j) {
        const int d = threadIdx.x + 256 * j;
        float t = x[d];
        if (Res) t += Res[(size_t)row * D_DIM + d];
        v[j] = t;
        s += t;
    }
    sdata[threadIdx.x] = s;
    __syncthreads();
    for (int o = 128; o > 0; o >>= 1) {
        if (threadIdx.x < o) sdata[threadIdx.x] += sdata[threadIdx.x + o];
        __syncthreads();
    }
    const float mu = sdata[0] * (1.0f / D_DIM);
    __syncthreads();
    float vs = 0.f;
#pragma unroll
    for (int j = 0; j < 3; ++j) { const float d = v[j] - mu; vs += d * d; }
    sdata[threadIdx.x] = vs;
    __syncthreads();
    for (int o = 128; o > 0; o >>= 1) {
        if (threadIdx.x < o) sdata[threadIdx.x] += sdata[threadIdx.x + o];
        __syncthreads();
    }
    const float rstd = rsqrtf(sdata[0] * (1.0f / D_DIM) + 1e-5f);
#pragma unroll
    for (int j = 0; j < 3; ++j) {
        const int d = threadIdx.x + 256 * j;
        Y[(size_t)row * D_DIM + d] = (v[j] - mu) * rstd * scale[d] + bias[d];
    }
}

// ---------------------------------------------------------------------------
// Tiled WMMA GEMM: C[M,N] = epi(A[M,K] @ W[K,N] + bias)
// Block tile 128x64, K-step 32, 256 threads = 8 waves; wave w owns rows
// [w*16, w*16+16) x all 64 cols -> 4 accumulators, 4 WMMAs per K-step.
// A staged f32 via async global->LDS (ASYNCcnt) when available; converted to
// f16 at fragment-gather. B staged f32->f16, K-transposed, so B-fragments are
// two 16B LDS reads per the CDNA5 16-bit B VGPR striping.
// EPI: 0 = none, 1 = +bias, 2 = +bias then exact GELU.
// M must be a multiple of 128 (true here); N may be ragged (logits).
// ---------------------------------------------------------------------------
template <int EPI>
__global__ __launch_bounds__(256) void gemm_kernel(const float* __restrict__ A,
                                                   const float* __restrict__ W,
                                                   const float* __restrict__ bias,
                                                   float* __restrict__ C,
                                                   int M, int N, int K) {
    __shared__ alignas(16) float    Asf[128][36];  // [m][k] f32, row 144B (16B mult)
    __shared__ alignas(16) _Float16 Bt[64][48];    // [n][k] f16, K-transposed

    const int tid   = threadIdx.x;
    const int lane  = tid & 31;
    const int wave  = tid >> 5;     // 0..7 -> 16-row strip
    const int m0    = blockIdx.x * 128;
    const int n0    = blockIdx.y * 64;
    const int l16   = lane & 15;
    const int khalf = lane >> 4;

    v8f acc[4] = {};

    for (int k0 = 0; k0 < K; k0 += 32) {
        // ---- stage A tile 128x32 f32 (async DMA to LDS when available) ----
#ifdef HAVE_ASYNC_LDS
#pragma unroll
        for (int j = 0; j < 4; ++j) {
            const int c = tid + 256 * j;          // 1024 16B chunks
            const int r = c >> 3, cc = (c & 7) * 4;
            async_copy16(&A[(size_t)(m0 + r) * K + (k0 + cc)], &Asf[r][cc]);
        }
#else
#pragma unroll
        for (int j = 0; j < 4; ++j) {
            const int c = tid + 256 * j;
            const int r = c >> 3, cc = (c & 7) * 4;
            *(float4*)&Asf[r][cc] = *(const float4*)&A[(size_t)(m0 + r) * K + (k0 + cc)];
        }
#endif
        // ---- stage B tile 32x64 transposed f16 (coalesced over n) ----
#pragma unroll
        for (int j = 0; j < 8; ++j) {
            const int i  = tid + 256 * j;
            const int kk = i >> 6, nn = i & 63;
            float w = 0.f;
            if (n0 + nn < N) w = W[(size_t)(k0 + kk) * N + (n0 + nn)];
            Bt[nn][kk] = (_Float16)w;
        }
        if (k0 + 32 < K) {  // pull next K-tile toward the caches
            __builtin_prefetch(&A[(size_t)(m0 + (tid >> 1)) * K + (k0 + 32)], 0, 1);
            __builtin_prefetch(&W[(size_t)(k0 + 32 + (tid >> 6)) * N + n0 + (tid & 63)], 0, 1);
        }
#ifdef HAVE_ASYNC_LDS
        async_wait0();
#endif
        __syncthreads();

        // ---- A fragment (f32 LDS -> f16 regs), CDNA5 16-bit A striping ----
        const float* ap = &Asf[wave * 16 + l16][khalf * 8];
        v16h a;
#pragma unroll
        for (int j = 0; j < 8; ++j) { a[j] = (_Float16)ap[j]; a[j + 8] = (_Float16)ap[j + 16]; }

        // ---- 4 sub-tiles of 16 cols each ----
#pragma unroll
        for (int t = 0; t < 4; ++t) {
            const int nr = t * 16 + l16;
            const v16h b = cat8(*(const v8h*)&Bt[nr][khalf * 8],
                                *(const v8h*)&Bt[nr][khalf * 8 + 16]);
            acc[t] = __builtin_amdgcn_wmma_f32_16x16x32_f16(false, a, false, b, (short)0, acc[t], false, false);
        }
        __syncthreads();
    }

    // ---- epilogue (C layout: lane->n=l16, vgpr r -> m = khalf*8 + r) ----
    const int row = m0 + wave * 16 + khalf * 8;
#pragma unroll
    for (int t = 0; t < 4; ++t) {
        const int col = n0 + t * 16 + l16;
        if (col < N) {
            const float bv = (EPI >= 1) ? bias[col] : 0.f;
#pragma unroll
            for (int r = 0; r < 8; ++r) {
                float v = acc[t][r] + bv;
                if (EPI == 2) v = gelu_exact(v);
                C[(size_t)(row + r) * N + col] = v;
            }
        }
    }
}

// ---------------------------------------------------------------------------
// Pack f32 q/k/v into f16: Qh,Kh keep [S,D]; V is transposed to Vt[h*64+hd][s]
// so PV B-fragments are contiguous.
// ---------------------------------------------------------------------------
__global__ __launch_bounds__(256) void pack_kernel(const float* __restrict__ q,
                                                   const float* __restrict__ k,
                                                   const float* __restrict__ v,
                                                   _Float16* __restrict__ Qh,
                                                   _Float16* __restrict__ Kh,
                                                   _Float16* __restrict__ Vt) {
    const int i = blockIdx.x * 256 + threadIdx.x;
    if (i >= S_LEN * D_DIM) return;
    Qh[i] = (_Float16)q[i];
    Kh[i] = (_Float16)k[i];
    const int s = i / D_DIM, d = i % D_DIM;   // d == h*HD + hd
    Vt[(size_t)d * S_LEN + s] = (_Float16)v[i];
}

// ---------------------------------------------------------------------------
// Attention: one wave per (16-query block, head).
//  pass1: E[16][1024] = QK^T/8 + mask   (WMMA, B-frag direct from Kh [S,D])
//  softmax in LDS (wave-local)
//  pass2: out = P @ V                   (WMMA, B-frag from transposed Vt)
// Dynamic LDS: 16*1024*4 = 64KB.
// ---------------------------------------------------------------------------
__global__ __launch_bounds__(32) void attn_kernel(const _Float16* __restrict__ Qh,
                                                  const _Float16* __restrict__ Kh,
                                                  const _Float16* __restrict__ Vt,
                                                  float* __restrict__ Out) {
    extern __shared__ float E[];   // [16][S_LEN]
    __shared__ float red[64];
    const int lane  = threadIdx.x;
    const int l16   = lane & 15;
    const int khalf = lane >> 4;
    const int h     = blockIdx.y;
    const int q0    = blockIdx.x * 16;

    // Q fragments (A-matrix 16x32, two K-slabs covering HD=64) stay in regs
    const _Float16* qp = Qh + (size_t)(q0 + l16) * D_DIM + h * HD;
    const v16h aq0 = cat8(*(const v8h*)(qp + khalf * 8),
                          *(const v8h*)(qp + khalf * 8 + 16));
    const v16h aq1 = cat8(*(const v8h*)(qp + 32 + khalf * 8),
                          *(const v8h*)(qp + 32 + khalf * 8 + 16));

    // ---- pass 1: scores ----
    for (int kt = 0; kt < S_LEN / 16; ++kt) {
        const int key0 = kt * 16;
        const _Float16* kp = Kh + (size_t)(key0 + l16) * D_DIM + h * HD;
        const v16h b0 = cat8(*(const v8h*)(kp + khalf * 8), *(const v8h*)(kp + khalf * 8 + 16));        // hd 0..31 slab
        const v16h b1 = cat8(*(const v8h*)(kp + 32 + khalf * 8), *(const v8h*)(kp + 32 + khalf * 8 + 16));   // hd 32..63 slab
        v8f e = {};
        e = __builtin_amdgcn_wmma_f32_16x16x32_f16(false, aq0, false, b0, (short)0, e, false, false);
        e = __builtin_amdgcn_wmma_f32_16x16x32_f16(false, aq1, false, b1, (short)0, e, false, false);
#pragma unroll
        for (int r = 0; r < 8; ++r) {
            const int ql  = khalf * 8 + r;
            const int key = key0 + l16;
            const float m = (key <= q0 + ql) ? 1.0f : -10000.0f;
            E[ql * S_LEN + key] = e[r] * 0.125f + m;   // 1/sqrt(64)
        }
    }
    __syncthreads();

    // ---- softmax: lane owns (row=l16, half=khalf of 512 cols) ----
    {
        float* rowp = E + l16 * S_LEN + khalf * 512;
        float mx = -3.4e38f;
        for (int i = 0; i < 512; ++i) mx = fmaxf(mx, rowp[i]);
        red[lane] = mx;
        __syncthreads();
        mx = fmaxf(red[l16], red[l16 + 16]);
        float sum = 0.f;
        for (int i = 0; i < 512; ++i) { const float p = __expf(rowp[i] - mx); rowp[i] = p; sum += p; }
        red[lane + 32] = sum;
        __syncthreads();
        const float inv = 1.0f / (red[l16 + 32] + red[l16 + 48]);
        for (int i = 0; i < 512; ++i) rowp[i] *= inv;
    }
    __syncthreads();

    // ---- pass 2: P @ V ----
    v8f acc[4] = {};
    for (int kc = 0; kc < S_LEN / 32; ++kc) {
        const float* pa = E + l16 * S_LEN + kc * 32 + khalf * 8;
        v16h a;
#pragma unroll
        for (int j = 0; j < 8; ++j) { a[j] = (_Float16)pa[j]; a[j + 8] = (_Float16)pa[j + 16]; }
#pragma unroll
        for (int t = 0; t < 4; ++t) {
            const _Float16* vp = Vt + (size_t)(h * HD + t * 16 + l16) * S_LEN + kc * 32 + khalf * 8;
            const v16h b = cat8(*(const v8h*)vp, *(const v8h*)(vp + 16));
            acc[t] = __builtin_amdgcn_wmma_f32_16x16x32_f16(false, a, false, b, (short)0, acc[t], false, false);
        }
    }
#pragma unroll
    for (int t = 0; t < 4; ++t)
#pragma unroll
        for (int r = 0; r < 8; ++r)
            Out[(size_t)(q0 + khalf * 8 + r) * D_DIM + h * HD + t * 16 + l16] = acc[t][r];
}

// ---------------------------------------------------------------------------
// Host-side orchestration
// ---------------------------------------------------------------------------
extern "C" void kernel_launch(void* const* d_in, const int* in_sizes, int n_in,
                              void* d_out, int out_size, void* d_ws, size_t ws_size,
                              hipStream_t stream) {
    (void)in_sizes; (void)n_in; (void)out_size; (void)ws_size;
    const int*   ids = (const int*)d_in[0];
    const float* wte = (const float*)d_in[1];
    const float* lbs = (const float*)d_in[2];
    const float* lbb = (const float*)d_in[3];
    const float* wq  = (const float*)d_in[4];
    const float* wk  = (const float*)d_in[5];
    const float* wv  = (const float*)d_in[6];
    const float* wo  = (const float*)d_in[7];
    const float* bo  = (const float*)d_in[8];
    const float* las = (const float*)d_in[9];
    const float* lab = (const float*)d_in[10];
    const float* w1  = (const float*)d_in[11];
    const float* b1  = (const float*)d_in[12];
    const float* w2  = (const float*)d_in[13];
    const float* b2  = (const float*)d_in[14];
    const float* fls = (const float*)d_in[15];
    const float* flb = (const float*)d_in[16];
    const float* wf  = (const float*)d_in[17];
    const float* bf  = (const float*)d_in[18];
    float* logits = (float*)d_out;

    const size_t SD = (size_t)S_LEN * D_DIM;
    const size_t SF = (size_t)S_LEN * F_DIM;
    float* ws   = (float*)d_ws;
    float* x    = ws + 0 * SD;
    float* xn   = ws + 1 * SD;
    float* q    = ws + 2 * SD;
    float* k    = ws + 3 * SD;
    float* v    = ws + 4 * SD;
    float* attn = ws + 5 * SD;
    float* o    = ws + 6 * SD;
    float* x2   = ws + 7 * SD;
    float* h1   = ws + 8 * SD;                 // [S, F]
    _Float16* Qh = (_Float16*)(ws + 8 * SD + SF);
    _Float16* Kh = Qh + SD;
    _Float16* Vt = Kh + SD;

    const dim3 blk256(256), blk32(32);
    const dim3 gemmDD(S_LEN / 128, D_DIM / 64);
    const dim3 gemmDF(S_LEN / 128, F_DIM / 64);
    const dim3 gemmDV(S_LEN / 128, (V_DIM + 63) / 64);
    const dim3 attnGrid(S_LEN / 16, H_NUM);
    const int  packBlocks = (int)((SD + 255) / 256);

    embed_kernel<<<S_LEN, blk256, 0, stream>>>(ids, wte, x);

    for (int l = 0; l < L_NUM; ++l) {
        const float* Wq  = wq + (size_t)l * D_DIM * D_DIM;
        const float* Wk  = wk + (size_t)l * D_DIM * D_DIM;
        const float* Wv  = wv + (size_t)l * D_DIM * D_DIM;
        const float* Wo  = wo + (size_t)l * D_DIM * D_DIM;
        const float* Bo  = bo + (size_t)l * D_DIM;
        const float* W1  = w1 + (size_t)l * D_DIM * F_DIM;
        const float* B1  = b1 + (size_t)l * F_DIM;
        const float* W2  = w2 + (size_t)l * F_DIM * D_DIM;
        const float* B2  = b2 + (size_t)l * D_DIM;
        const float* Lbs = lbs + (size_t)l * D_DIM;
        const float* Lbb = lbb + (size_t)l * D_DIM;
        const float* Las = las + (size_t)l * D_DIM;
        const float* Lab = lab + (size_t)l * D_DIM;

        // xn = LN(x)
        ln_kernel<<<S_LEN, blk256, 0, stream>>>(x, nullptr, Lbs, Lbb, xn);
        // q/k/v = xn @ W{q,k,v}   (no bias in reference)
        gemm_kernel<0><<<gemmDD, blk256, 0, stream>>>(xn, Wq, nullptr, q, S_LEN, D_DIM, D_DIM);
        gemm_kernel<0><<<gemmDD, blk256, 0, stream>>>(xn, Wk, nullptr, k, S_LEN, D_DIM, D_DIM);
        gemm_kernel<0><<<gemmDD, blk256, 0, stream>>>(xn, Wv, nullptr, v, S_LEN, D_DIM, D_DIM);
        // f16 pack + V transpose
        pack_kernel<<<packBlocks, blk256, 0, stream>>>(q, k, v, Qh, Kh, Vt);
        // attention (64KB dynamic LDS score buffer)
        attn_kernel<<<attnGrid, blk32, 16 * S_LEN * sizeof(float), stream>>>(Qh, Kh, Vt, attn);
        // out projection + bias
        gemm_kernel<1><<<gemmDD, blk256, 0, stream>>>(attn, Wo, Bo, o, S_LEN, D_DIM, D_DIM);
        // post-LN residual:  x2 = LN(o + xn)
        ln_kernel<<<S_LEN, blk256, 0, stream>>>(o, xn, Las, Lab, x2);
        // MLP: gelu(x2@W1 + B1) then gelu(h1@W2 + B2); no residual (per source)
        gemm_kernel<2><<<gemmDF, blk256, 0, stream>>>(x2, W1, B1, h1, S_LEN, F_DIM, D_DIM);
        gemm_kernel<2><<<gemmDD, blk256, 0, stream>>>(h1, W2, B2, x, S_LEN, D_DIM, F_DIM);
    }

    // final LN + logits
    ln_kernel<<<S_LEN, blk256, 0, stream>>>(x, nullptr, fls, flb, xn);
    gemm_kernel<1><<<gemmDV, blk256, 0, stream>>>(xn, wf, bf, logits, S_LEN, V_DIM, D_DIM);
}
